// GINBackboneOGB_33921651703942
// MI455X (gfx1250) — matrix operand order repaired and checked by reading, not compiled, not run on hardware
//
#include <hip/hip_runtime.h>

// ---------------------------------------------------------------------------
// GIN backbone for MI455X (gfx1250). GEMMs run on v_wmma_f32_16x16x32_bf16.
// Dims padded: K 300->320, hidden 600->640, out 300->320 (zero-filled pads).
// Workspace layout (<= ~260.2 MB):
//   agg/hn (aliased) : N*320*4  = 128,000,000 B
//   Hid bf16         : N*640*2  = 128,000,000 B
//   W1t bf16         : 5*640*320*2 = 2,048,000 B
//   W2t bf16         : 5*320*640*2 = 2,048,000 B
//   stats            : 4*320*4 floats (sums, sqsums, scale, shift)
// d_out ([N,300] f32) is reused as the persistent node-feature buffer h.
// ---------------------------------------------------------------------------

#define N_NODES 100000
#define N_EDGES 200000
#define EMB     300
#define DK      320   // padded K for GEMM1 / out of GEMM2
#define HID     640   // padded hidden (out of GEMM1 / K of GEMM2)
#define NLAYER  5

typedef __attribute__((ext_vector_type(16))) __bf16 v16bf;
typedef __attribute__((ext_vector_type(8)))  float  v8f;

union Frag { uint4 u[2]; v16bf v; };

// ---------------------------------------------------------------------------
// Atom-embedding sum: h[n,:] = sum_i atom_tab[i, x[n,i], :]
// ---------------------------------------------------------------------------
__global__ __launch_bounds__(256) void emb_kernel(const float* __restrict__ atom_tab,
                                                  const int* __restrict__ x,
                                                  float* __restrict__ h) {
  int t = blockIdx.x * 256 + threadIdx.x;
  if (t >= N_NODES * 75) return;
  int n = t / 75, c4 = t % 75;
  float4 acc = {0.f, 0.f, 0.f, 0.f};
#pragma unroll
  for (int i = 0; i < 9; ++i) {
    int v = x[n * 9 + i];
    float4 p = *(const float4*)(atom_tab + (size_t)(i * 120 + v) * EMB + c4 * 4);
    acc.x += p.x; acc.y += p.y; acc.z += p.z; acc.w += p.w;
  }
  *(float4*)(h + (size_t)n * EMB + c4 * 4) = acc;
}

// ---------------------------------------------------------------------------
// One-time: transpose + cast weights to bf16, zero-padded.
// W1t[l][n<640][k<320] = W1[l][k][n];  W2t[l][n<320][k<640] = W2[l][k][n]
// ---------------------------------------------------------------------------
__global__ __launch_bounds__(256) void wconv_kernel(const float* __restrict__ W1,
                                                    const float* __restrict__ W2,
                                                    __bf16* __restrict__ W1t,
                                                    __bf16* __restrict__ W2t) {
  const int HALF = NLAYER * HID * DK;  // == NLAYER * DK * HID
  int t = blockIdx.x * 256 + threadIdx.x;
  if (t < HALF) {
    int l = t / (HID * DK), r = t % (HID * DK);
    int n = r / DK, k = r % DK;
    float v = (n < 600 && k < 300) ? W1[((size_t)l * 300 + k) * 600 + n] : 0.f;
    W1t[t] = (__bf16)v;
  } else if (t < 2 * HALF) {
    int t2 = t - HALF;
    int l = t2 / (DK * HID), r = t2 % (DK * HID);
    int n = r / HID, k = r % HID;
    float v = (n < 300 && k < 600) ? W2[((size_t)l * 600 + k) * 300 + n] : 0.f;
    W2t[t2] = (__bf16)v;
  }
}

// ---------------------------------------------------------------------------
// agg[n, c] = h[n, c] + e_self[c] (c<300), 0 for pad cols. Self-loop attr=[4,0,0].
// ---------------------------------------------------------------------------
__global__ __launch_bounds__(256) void agg_init_kernel(const float* __restrict__ h,
                                                       const float* __restrict__ bt_l,
                                                       float* __restrict__ agg) {
  int t = blockIdx.x * 256 + threadIdx.x;
  if (t >= N_NODES * 80) return;
  int n = t / 80, c4 = t % 80;
  float4 v = {0.f, 0.f, 0.f, 0.f};
  if (c4 < 75) {
    int c = c4 * 4;
    float4 hv = *(const float4*)(h + (size_t)n * EMB + c);
    float4 b0 = *(const float4*)(bt_l + (size_t)(0 * 6 + 4) * EMB + c);
    float4 b1 = *(const float4*)(bt_l + (size_t)(1 * 6 + 0) * EMB + c);
    float4 b2 = *(const float4*)(bt_l + (size_t)(2 * 6 + 0) * EMB + c);
    v.x = hv.x + b0.x + b1.x + b2.x;
    v.y = hv.y + b0.y + b1.y + b2.y;
    v.z = hv.z + b0.z + b1.z + b2.z;
    v.w = hv.w + b0.w + b1.w + b2.w;
  }
  *(float4*)(agg + (size_t)n * DK + c4 * 4) = v;
}

// ---------------------------------------------------------------------------
// Per-edge message + scatter-add:  agg[dst] += h[src] + bond_emb(edge_attr)
// f32 atomics resolve in the 192MB L2 (agg = 128MB fits).
// ---------------------------------------------------------------------------
__global__ __launch_bounds__(256) void edge_scatter_kernel(const float* __restrict__ h,
                                                           const float* __restrict__ bt_l,
                                                           const int* __restrict__ ei,
                                                           const int* __restrict__ ea,
                                                           float* __restrict__ agg) {
  int t = blockIdx.x * 256 + threadIdx.x;
  if (t >= N_EDGES * 75) return;
  int e = t / 75, c = (t % 75) * 4;
  int src = ei[e], dst = ei[N_EDGES + e];
  int a0 = ea[e * 3 + 0], a1 = ea[e * 3 + 1], a2 = ea[e * 3 + 2];
  float4 hv = *(const float4*)(h + (size_t)src * EMB + c);
  float4 b0 = *(const float4*)(bt_l + (size_t)(0 * 6 + a0) * EMB + c);
  float4 b1 = *(const float4*)(bt_l + (size_t)(1 * 6 + a1) * EMB + c);
  float4 b2 = *(const float4*)(bt_l + (size_t)(2 * 6 + a2) * EMB + c);
  float* o = agg + (size_t)dst * DK + c;
  atomicAdd(o + 0, hv.x + b0.x + b1.x + b2.x);
  atomicAdd(o + 1, hv.y + b0.y + b1.y + b2.y);
  atomicAdd(o + 2, hv.z + b0.z + b1.z + b2.z);
  atomicAdd(o + 3, hv.w + b0.w + b1.w + b2.w);
}

// ---------------------------------------------------------------------------
// WMMA GEMM: Out[M x Nout] = A[M x K] * Bt[Nout x K]^T (+bias, opt relu)
// block = 256 threads = 8 wave32 as 4(M) x 2(N); wave tile 32x32 (2x2 WMMA).
// Block tile: 128(M) x 64(N). LDS tiles padded (+8 bf16) vs bank conflicts.
// ---------------------------------------------------------------------------
template<int K, int LDO, int BIASN, bool A_F32, bool RELU, bool OUT_BF16>
__global__ __launch_bounds__(256) void gemm_wmma_kernel(const void* __restrict__ Ap,
                                                        const __bf16* __restrict__ Bt,
                                                        const float* __restrict__ bias,
                                                        void* __restrict__ Outp, int M) {
  __shared__ __bf16 sA[128][40];
  __shared__ __bf16 sB[64][40];
  const int tid  = threadIdx.x;
  const int lane = tid & 31, wave = tid >> 5;
  const int wm = wave >> 1, wn = wave & 1;
  const int lr = lane & 15, hi = lane >> 4;
  const int mBase = blockIdx.x * 128;
  const int nBase = blockIdx.y * 64;

  v8f acc[2][2] = {};

  for (int kb = 0; kb < K; kb += 32) {
    // ---- stage A tile [128 x 32] (cast f32->bf16 in flight for GEMM1) ----
    if constexpr (A_F32) {
      const float* A = (const float*)Ap;
#pragma unroll
      for (int i = 0; i < 4; ++i) {
        int e = tid + i * 256;           // 1024 float4 elements
        int row = e >> 3, c4 = e & 7;
        int gm = mBase + row; if (gm >= M) gm = M - 1;
        float4 f = *(const float4*)(A + (size_t)gm * K + kb + c4 * 4);
        __bf16* d = &sA[row][c4 * 4];
        d[0] = (__bf16)f.x; d[1] = (__bf16)f.y; d[2] = (__bf16)f.z; d[3] = (__bf16)f.w;
        if (kb + 32 < K)                 // global_prefetch of next K tile
          __builtin_prefetch(A + (size_t)gm * K + kb + 32 + c4 * 4, 0, 0);
      }
    } else {
      const __bf16* A = (const __bf16*)Ap;
#pragma unroll
      for (int i = 0; i < 2; ++i) {
        int e = tid + i * 256;           // 512 x (8 bf16)
        int row = e >> 2, c8 = e & 3;
        int gm = mBase + row; if (gm >= M) gm = M - 1;
        *(uint4*)&sA[row][c8 * 8] = *(const uint4*)(A + (size_t)gm * K + kb + c8 * 8);
      }
    }
    // ---- stage B tile [64 x 32] from pre-transposed weights ----
    {
      int row = tid >> 2, c8 = tid & 3;
      *(uint4*)&sB[row][c8 * 8] = *(const uint4*)(Bt + (size_t)(nBase + row) * K + kb + c8 * 8);
    }
    __syncthreads();

    // ---- fragments per ISA 16-bit A/B wave32 layouts ----
    Frag fa[2], fb[2];
#pragma unroll
    for (int tm = 0; tm < 2; ++tm) {
      int r = wm * 32 + tm * 16 + lr;
      int kh = hi * 8;                           // lanes 0-15: K0-7/16-23; 16-31: K8-15/24-31
      fa[tm].u[0] = *(const uint4*)&sA[r][kh];
      fa[tm].u[1] = *(const uint4*)&sA[r][kh + 16];
    }
#pragma unroll
    for (int tn = 0; tn < 2; ++tn) {
      int n = wn * 32 + tn * 16 + lr;
      int kk = hi * 8;                           // lanes 0-15: K0-7/16-23; 16-31: K8-15/24-31
      fb[tn].u[0] = *(const uint4*)&sB[n][kk];
      fb[tn].u[1] = *(const uint4*)&sB[n][kk + 16];
    }
#pragma unroll
    for (int tm = 0; tm < 2; ++tm)
#pragma unroll
      for (int tn = 0; tn < 2; ++tn)
        acc[tm][tn] = __builtin_amdgcn_wmma_f32_16x16x32_bf16(
            false, fa[tm].v, false, fb[tn].v, (short)0, acc[tm][tn], false, false);
    __syncthreads();
  }

  // ---- epilogue: bias (+relu), store (C/D layout: vgpr r -> row, lane -> col) ----
#pragma unroll
  for (int tn = 0; tn < 2; ++tn) {
    int ng = nBase + wn * 32 + tn * 16 + lr;
    float bv = (ng < BIASN) ? bias[ng] : 0.f;
#pragma unroll
    for (int tm = 0; tm < 2; ++tm) {
#pragma unroll
      for (int r = 0; r < 8; ++r) {
        int mg = mBase + wm * 32 + tm * 16 + hi * 8 + r;
        if (mg < M) {
          float v = acc[tm][tn][r] + bv;
          if (RELU && v < 0.f) v = 0.f;
          if constexpr (OUT_BF16) ((__bf16*)Outp)[(size_t)mg * LDO + ng] = (__bf16)v;
          else                    ((float*)Outp)[(size_t)mg * LDO + ng] = v;
        }
      }
    }
  }
}

// ---------------------------------------------------------------------------
// BatchNorm: column sums/sumsq -> scale/shift -> apply (+relu except last)
// ---------------------------------------------------------------------------
__global__ void zero_stats_kernel(float* __restrict__ stats) {
  int t = blockIdx.x * 256 + threadIdx.x;
  if (t < 640) stats[t] = 0.f;  // sums[320] + sqsums[320]
}

__global__ __launch_bounds__(320) void bn_stats_kernel(const float* __restrict__ hn,
                                                       float* __restrict__ stats) {
  int c = threadIdx.x;
  if (c >= EMB) return;
  int r0 = blockIdx.x * 256;
  int r1 = r0 + 256; if (r1 > N_NODES) r1 = N_NODES;
  float s = 0.f, q = 0.f;
  for (int r = r0; r < r1; ++r) {
    float v = hn[(size_t)r * DK + c];
    s += v; q += v * v;
  }
  atomicAdd(&stats[c], s);
  atomicAdd(&stats[320 + c], q);
}

__global__ __launch_bounds__(320) void bn_final_kernel(const float* __restrict__ stats,
                                                       const float* __restrict__ gamma,
                                                       const float* __restrict__ beta,
                                                       float* __restrict__ scale_shift) {
  int c = threadIdx.x;
  if (c >= EMB) return;
  float mean = stats[c] * (1.f / N_NODES);
  float var  = stats[320 + c] * (1.f / N_NODES) - mean * mean;
  float sc = gamma[c] * rsqrtf(var + 1e-5f);
  scale_shift[c]       = sc;
  scale_shift[320 + c] = beta[c] - mean * sc;
}

__global__ __launch_bounds__(256) void bn_apply_kernel(const float* __restrict__ hn,
                                                       const float* __restrict__ scale_shift,
                                                       float* __restrict__ h_out, int relu) {
  int t = blockIdx.x * 256 + threadIdx.x;
  if (t >= N_NODES * 75) return;
  int n = t / 75, c = (t % 75) * 4;
  float4 v  = *(const float4*)(hn + (size_t)n * DK + c);
  float4 sc = *(const float4*)(scale_shift + c);
  float4 sh = *(const float4*)(scale_shift + 320 + c);
  v.x = v.x * sc.x + sh.x; v.y = v.y * sc.y + sh.y;
  v.z = v.z * sc.z + sh.z; v.w = v.w * sc.w + sh.w;
  if (relu) {
    v.x = v.x < 0.f ? 0.f : v.x; v.y = v.y < 0.f ? 0.f : v.y;
    v.z = v.z < 0.f ? 0.f : v.z; v.w = v.w < 0.f ? 0.f : v.w;
  }
  *(float4*)(h_out + (size_t)n * EMB + c) = v;
}

// ---------------------------------------------------------------------------
extern "C" void kernel_launch(void* const* d_in, const int* in_sizes, int n_in,
                              void* d_out, int out_size, void* d_ws, size_t ws_size,
                              hipStream_t stream) {
  const float* atom_tab = (const float*)d_in[0];
  const float* bond_tab = (const float*)d_in[1];
  const float* W1  = (const float*)d_in[2];
  const float* b1  = (const float*)d_in[3];
  const float* W2  = (const float*)d_in[4];
  const float* b2  = (const float*)d_in[5];
  const float* gma = (const float*)d_in[6];
  const float* bta = (const float*)d_in[7];
  const int*   x   = (const int*)d_in[8];
  const int*   ei  = (const int*)d_in[9];
  const int*   ea  = (const int*)d_in[10];
  float* h = (float*)d_out;  // persistent node features [N, 300]

  char* w = (char*)d_ws;
  float*  agg  = (float*)w;                       // [N, 320] f32, aliased as hn
  __bf16* Hid  = (__bf16*)(w + 128000000);        // [N, 640] bf16
  __bf16* W1t  = (__bf16*)(w + 256000000);        // [5][640][320] bf16
  __bf16* W2t  = (__bf16*)(w + 258048000);        // [5][320][640] bf16
  float*  stats = (float*)(w + 260096000);        // sums/sqsums [640]
  float*  ssh   = stats + 640;                    // scale/shift [640]
  float*  hn   = agg;                             // GEMM2 output reuses agg

  const int gEmb   = (N_NODES * 75 + 255) / 256;
  const int gWcv   = (2 * NLAYER * HID * DK + 255) / 256;
  const int gInit  = (N_NODES * 80 + 255) / 256;
  const int gEdge  = (N_EDGES * 75 + 255) / 256;
  const int gStats = (N_NODES + 255) / 256;
  const int gMM    = (N_NODES + 127) / 128;

  emb_kernel<<<gEmb, 256, 0, stream>>>(atom_tab, x, h);
  wconv_kernel<<<gWcv, 256, 0, stream>>>(W1, W2, W1t, W2t);

  for (int l = 0; l < NLAYER; ++l) {
    const float* bt_l = bond_tab + (size_t)l * 4 * 6 * EMB;
    zero_stats_kernel<<<3, 256, 0, stream>>>(stats);
    agg_init_kernel<<<gInit, 256, 0, stream>>>(h, bt_l, agg);
    edge_scatter_kernel<<<gEdge, 256, 0, stream>>>(h, bt_l, ei, ea, agg);
    // GEMM1: Hid = relu(agg @ W1 + b1)   [N,320]x[320,640] -> bf16 [N,640]
    gemm_wmma_kernel<DK, HID, 600, true, true, true>
        <<<dim3(gMM, HID / 64), 256, 0, stream>>>(agg, W1t + (size_t)l * HID * DK,
                                                  b1 + (size_t)l * 600, Hid, N_NODES);
    // GEMM2: hn = Hid @ W2 + b2          [N,640]x[640,320] -> f32 [N,320]
    gemm_wmma_kernel<HID, DK, 300, false, false, false>
        <<<dim3(gMM, DK / 64), 256, 0, stream>>>(Hid, W2t + (size_t)l * DK * HID,
                                                 b2 + (size_t)l * 300, hn, N_NODES);
    bn_stats_kernel<<<gStats, 320, 0, stream>>>(hn, stats);
    bn_final_kernel<<<1, 320, 0, stream>>>(stats, gma + (size_t)l * EMB,
                                           bta + (size_t)l * EMB, ssh);
    bn_apply_kernel<<<gEmb, 256, 0, stream>>>(hn, ssh, h, l != NLAYER - 1 ? 1 : 0);
  }
}
